// Multiheadattention_45664092291236
// MI455X (gfx1250) — matrix operand order repaired and checked by reading, not compiled, not run on hardware
//
#include <hip/hip_runtime.h>
#include <hip/hip_bf16.h>

// Problem constants (match reference)
#define BB   8
#define SS   2048
#define DD   512
#define HH   8
#define LATD 64
#define OUTD 64
#define HOUT (HH * OUTD)   // 512

// ---------------------------------------------------------------------------
// WMMA fragment types (gfx1250, wave32)
// ---------------------------------------------------------------------------
typedef __attribute__((ext_vector_type(16))) __bf16       v16bf;
typedef __attribute__((ext_vector_type(8)))  float        v8f;
typedef __attribute__((ext_vector_type(8)))  unsigned int v8u;

union frag_u {
    v16bf b;
    v8u   u;
};

__device__ __forceinline__ v8f bf16_wmma(const frag_u& a, const frag_u& b, v8f c) {
    // D = A(16x32 bf16) * B(32x16 bf16) + C(16x16 f32)
    return __builtin_amdgcn_wmma_f32_16x16x32_bf16(
        /*neg_a=*/false, a.b, /*neg_b=*/false, b.b,
        /*c_mod=*/(short)0, c, /*reuse_a=*/false, /*reuse_b=*/false);
}

__device__ __forceinline__ unsigned short f32_to_bf16(float f) {
    unsigned int u = __builtin_bit_cast(unsigned int, f);
    u += 0x7FFFu + ((u >> 16) & 1u);   // round-to-nearest-even
    return (unsigned short)(u >> 16);
}

// Row reductions across the 16 lanes of a half-wave (xor masks never touch bit 4)
__device__ __forceinline__ float rowmax16(float v) {
#pragma unroll
    for (int m = 1; m < 16; m <<= 1) v = fmaxf(v, __shfl_xor(v, m, 32));
    return v;
}
__device__ __forceinline__ float rowsum16(float v) {
#pragma unroll
    for (int m = 1; m < 16; m <<= 1) v += __shfl_xor(v, m, 32);
    return v;
}

// A-fragment (16x32, 16-bit) K offset for element pair starting at e=2v (contiguous: k = 16*half + e)
__device__ __forceinline__ int a_koff(int v, int half) {
    int e = 2 * v;
    return e + 16 * half;
}

// Per-lane async copy: 16 bytes global -> LDS, tracked by ASYNCcnt.
__device__ __forceinline__ void async_copy_b128(const void* lds_dst, const void* gsrc) {
    unsigned int lds_off = (unsigned int)(unsigned long long)lds_dst;  // wave-relative LDS byte offset
    asm volatile("global_load_async_to_lds_b128 %0, %1, off"
                 :
                 : "v"(lds_off), "v"(gsrc)
                 : "memory");
}

// ---------------------------------------------------------------------------
// Kernel 0a: pack x (f32) -> bf16 pairs
// ---------------------------------------------------------------------------
__global__ void __launch_bounds__(256) cvt_x_kernel(const float* __restrict__ x,
                                                    unsigned int* __restrict__ xb,
                                                    int n2) {
    int i = blockIdx.x * 256 + threadIdx.x;
    if (i < n2) {
        float a = x[2 * i + 0];
        float b = x[2 * i + 1];
        xb[i] = (unsigned int)f32_to_bf16(a) | ((unsigned int)f32_to_bf16(b) << 16);
    }
}

// ---------------------------------------------------------------------------
// Kernel 0b: transpose + convert weight: in[batch][rows][cols] f32 ->
//            out[batch][cols][rows] bf16  (K-contiguous for B-fragments)
// ---------------------------------------------------------------------------
__global__ void __launch_bounds__(256) transpose_bf16_kernel(const float* __restrict__ in,
                                                             unsigned short* __restrict__ out,
                                                             int batch, int rows, int cols) {
    int i = blockIdx.x * 256 + threadIdx.x;
    int total = batch * rows * cols;
    if (i < total) {
        int c = i % cols;
        int t = i / cols;
        int r = t % rows;
        int bi = t / rows;
        out[((size_t)bi * cols + c) * rows + r] = f32_to_bf16(in[i]);
    }
}

// ---------------------------------------------------------------------------
// Kernel 1: QKV projections.
//   Q[b,h,s,lat] = x[b,s,:] @ WQ[h]   (bf16 out, row-major)
//   K  same layout;  V written transposed Vt[b,h,o,s]
// One wave = 16 rows of s, all 64 output cols, for Q,K,V. 8 waves/block.
// ---------------------------------------------------------------------------
__global__ void __launch_bounds__(256) proj_kernel(
    const unsigned short* __restrict__ xb,   // [B][S][D] bf16
    const unsigned short* __restrict__ wqt,  // [H][LAT][D] bf16
    const unsigned short* __restrict__ wkt,  // [H][LAT][D]
    const unsigned short* __restrict__ wvt,  // [H][OUT][D]
    unsigned short* __restrict__ Q,          // [B][H][S][LAT]
    unsigned short* __restrict__ Kb,         // [B][H][S][LAT]
    unsigned short* __restrict__ Vt)         // [B][H][OUT][S]
{
    const int bh   = blockIdx.x;
    const int b    = bh / HH;
    const int h    = bh % HH;
    const int wave = threadIdx.x >> 5;
    const int lane = threadIdx.x & 31;
    const int half = lane >> 4;
    const int l15  = lane & 15;
    const int s0   = blockIdx.y * 128 + wave * 16;

    const unsigned short* xrow = xb + ((size_t)(b * SS + s0)) * DD;
    __builtin_prefetch(xrow + (size_t)l15 * DD, 0, 0);   // global_prefetch_b8

    for (int mat = 0; mat < 3; ++mat) {
        const unsigned short* wt =
            (mat == 0 ? wqt : (mat == 1 ? wkt : wvt)) + (size_t)h * 64 * DD;

        v8f acc[4] = {};
        for (int kb = 0; kb < DD / 32; ++kb) {
            const int d0 = kb * 32;
            frag_u A;
#pragma unroll
            for (int v = 0; v < 8; ++v)
                A.u[v] = *(const unsigned int*)(xrow + (size_t)l15 * DD + d0 + a_koff(v, half));
#pragma unroll
            for (int nt = 0; nt < 4; ++nt) {
                frag_u Bf;
#pragma unroll
                for (int v = 0; v < 8; ++v)
                    Bf.u[v] = *(const unsigned int*)(wt + (size_t)(nt * 16 + l15) * DD +
                                                     d0 + 2 * v + 16 * half);
                acc[nt] = bf16_wmma(A, Bf, acc[nt]);
            }
        }
        // store D tiles (C/D layout: VGPR r -> row r+8*half, col = nt*16 + l15)
        if (mat < 2) {
            unsigned short* dst = (mat == 0 ? Q : Kb) + ((size_t)(b * HH + h) * SS) * LATD;
#pragma unroll
            for (int nt = 0; nt < 4; ++nt)
#pragma unroll
                for (int r = 0; r < 8; ++r)
                    dst[(size_t)(s0 + r + 8 * half) * LATD + nt * 16 + l15] =
                        f32_to_bf16(acc[nt][r]);
        } else {
            // V transposed: Vt[o][s]
#pragma unroll
            for (int nt = 0; nt < 4; ++nt) {
                unsigned short* dst =
                    Vt + ((size_t)(b * HH + h) * OUTD + nt * 16 + l15) * SS + s0 + 8 * half;
#pragma unroll
                for (int r = 0; r < 8; ++r) dst[r] = f32_to_bf16(acc[nt][r]);
            }
        }
    }
}

// ---------------------------------------------------------------------------
// Kernel 2: flash attention. Workgroup (8 waves) owns one (b,h) and 128
// queries; each wave owns 16 of them. Keys stream in blocks of 32 through a
// double-buffered LDS stage filled with global_load_async_to_lds_b128
// (ASYNCcnt), so the 8 waves share one copy of K/V per block. Scores & PV via
// WMMA; online softmax in exp2 domain; P reshaped D-layout -> A-fragment
// through per-wave LDS.
// ---------------------------------------------------------------------------
__global__ void __launch_bounds__(256) flash_kernel(
    const unsigned short* __restrict__ Q,    // [B][H][S][LAT] bf16
    const unsigned short* __restrict__ Kb,   // [B][H][S][LAT] bf16
    const unsigned short* __restrict__ Vt,   // [B][H][OUT][S] bf16
    unsigned short* __restrict__ Ocat)       // [B][S][H*OUT]  bf16
{
    // K tile: 32 keys x 64 lat; row padded 64->72 (144 B = 9*16 B, conflict-free)
    // V tile: 64 out  x 32 key; row padded 32->40 ( 80 B = 5*16 B, conflict-free)
    __shared__ unsigned short Kt[2][32][72];
    __shared__ unsigned short Vtl[2][64][40];
    __shared__ unsigned short Pb[8][16][40];   // per-wave 16x32 P tile, padded

    const int tid  = threadIdx.x;
    const int bh   = blockIdx.x;
    const int b    = bh / HH;
    const int h    = bh % HH;
    const int wave = tid >> 5;
    const int lane = tid & 31;
    const int half = lane >> 4;
    const int l15  = lane & 15;
    const int sq   = blockIdx.y * 128 + wave * 16;

    const unsigned short* Qp = Q  + ((size_t)(b * HH + h) * SS + sq) * LATD;
    const unsigned short* Kp = Kb + ((size_t)(b * HH + h) * SS) * LATD;
    const unsigned short* Vp = Vt + ((size_t)(b * HH + h) * OUTD) * SS;

    // cooperative async staging: each of the 256 threads moves 16 B per tile
    const int krow = tid >> 3, kseg = (tid & 7) * 8;   // K: 32 rows x 128 B
    const int vrow = tid >> 2, vseg = (tid & 3) * 8;   // V: 64 rows x  64 B

    // Preload Q A-fragments (lat 0..31 and 32..63) into registers
    frag_u qa[2];
#pragma unroll
    for (int kb = 0; kb < 2; ++kb)
#pragma unroll
        for (int v = 0; v < 8; ++v)
            qa[kb].u[v] =
                *(const unsigned int*)(Qp + (size_t)l15 * LATD + 32 * kb + a_koff(v, half));

    v8f   O[4] = {};
    float mrow[8], lrow[8];
#pragma unroll
    for (int r = 0; r < 8; ++r) { mrow[r] = -1e30f; lrow[r] = 0.0f; }

    const float sc = 0.125f * 1.44269504088896340736f;  // 1/sqrt(64) * log2(e)

    // prologue: stage first key block into buffer 0
    async_copy_b128(&Kt[0][krow][kseg], Kp + (size_t)krow * LATD + kseg);
    async_copy_b128(&Vtl[0][vrow][vseg], Vp + (size_t)vrow * SS + vseg);

    for (int j = 0; j < SS; j += 32) {
        const int cur = (j >> 5) & 1;
        if (j + 32 < SS) {
            // stage next block into the other buffer (stays in flight)
            async_copy_b128(&Kt[cur ^ 1][krow][kseg],
                            Kp + (size_t)(j + 32 + krow) * LATD + kseg);
            async_copy_b128(&Vtl[cur ^ 1][vrow][vseg],
                            Vp + (size_t)vrow * SS + j + 32 + vseg);
            asm volatile("s_wait_asynccnt 0x2" ::: "memory");  // current buf done
        } else {
            asm volatile("s_wait_asynccnt 0x0" ::: "memory");
        }
        __syncthreads();   // all waves' staged data visible

        // ---- scores: S0 = Q·K^T (keys j..j+15), S1 = keys j+16..j+31
        v8f s0 = {}, s1 = {};
#pragma unroll
        for (int kb = 0; kb < 2; ++kb) {
            frag_u B0, B1;
#pragma unroll
            for (int v = 0; v < 8; ++v) {
                const int kk = 2 * v + 16 * half + 32 * kb;
                B0.u[v] = *(const unsigned int*)(&Kt[cur][l15][kk]);
                B1.u[v] = *(const unsigned int*)(&Kt[cur][16 + l15][kk]);
            }
            s0 = bf16_wmma(qa[kb], B0, s0);
            s1 = bf16_wmma(qa[kb], B1, s1);
        }
        // ---- online softmax (per query row r+8*half)
#pragma unroll
        for (int r = 0; r < 8; ++r) {
            const float t0 = s0[r] * sc;
            const float t1 = s1[r] * sc;
            float mx = rowmax16(fmaxf(t0, t1));
            const float mn    = fmaxf(mrow[r], mx);
            const float alpha = __builtin_amdgcn_exp2f(mrow[r] - mn);
            const float p0    = __builtin_amdgcn_exp2f(t0 - mn);
            const float p1    = __builtin_amdgcn_exp2f(t1 - mn);
            lrow[r] = lrow[r] * alpha + rowsum16(p0 + p1);
            mrow[r] = mn;
#pragma unroll
            for (int nt = 0; nt < 4; ++nt) O[nt][r] *= alpha;
            Pb[wave][r + 8 * half][l15]      = f32_to_bf16(p0);
            Pb[wave][r + 8 * half][16 + l15] = f32_to_bf16(p1);
        }
        // same-wave LDS RAW: DS pipe is in-order per wave, wait to be explicit
        asm volatile("s_wait_dscnt 0" ::: "memory");

        // ---- reload P as A-fragment (16x32)
        frag_u pa;
#pragma unroll
        for (int v = 0; v < 8; ++v)
            pa.u[v] = *(const unsigned int*)(&Pb[wave][l15][a_koff(v, half)]);

        // ---- O += P · V (B-fragments from staged V tile)
#pragma unroll
        for (int nt = 0; nt < 4; ++nt) {
            frag_u Bv;
#pragma unroll
            for (int v = 0; v < 8; ++v)
                Bv.u[v] = *(const unsigned int*)(&Vtl[cur][nt * 16 + l15][2 * v + 16 * half]);
            O[nt] = bf16_wmma(pa, Bv, O[nt]);
        }
        __syncthreads();   // everyone done reading buf[cur] before it is refilled
    }

    // ---- epilogue: normalize and write concatenated heads (bf16)
#pragma unroll
    for (int r = 0; r < 8; ++r) {
        const float inv  = 1.0f / lrow[r];
        const int   srow = sq + r + 8 * half;
        unsigned short* orow = Ocat + ((size_t)b * SS + srow) * HOUT + h * OUTD;
#pragma unroll
        for (int nt = 0; nt < 4; ++nt)
            orow[nt * 16 + l15] = f32_to_bf16(O[nt][r] * inv);
    }
}

// ---------------------------------------------------------------------------
// Kernel 3: output projection  out[BS][64] = Ocat[BS][512] @ WO[512][64]
// ---------------------------------------------------------------------------
__global__ void __launch_bounds__(256) outproj_kernel(
    const unsigned short* __restrict__ Ocat,  // [B*S][512] bf16
    const unsigned short* __restrict__ wot,   // [64][512]  bf16 (WO^T)
    float* __restrict__ out)                  // [B*S][64]  f32
{
    const int mt   = blockIdx.x * 8 + (threadIdx.x >> 5);
    const int lane = threadIdx.x & 31;
    const int half = lane >> 4;
    const int l15  = lane & 15;
    const int row0 = mt * 16;

    __builtin_prefetch(Ocat + (size_t)(row0 + l15) * HOUT, 0, 0);  // global_prefetch_b8

    v8f acc[4] = {};
    for (int kb = 0; kb < HOUT / 32; ++kb) {
        const int d0 = kb * 32;
        frag_u A;
#pragma unroll
        for (int v = 0; v < 8; ++v)
            A.u[v] = *(const unsigned int*)(Ocat + (size_t)(row0 + l15) * HOUT + d0 +
                                            a_koff(v, half));
#pragma unroll
        for (int nt = 0; nt < 4; ++nt) {
            frag_u Bf;
#pragma unroll
            for (int v = 0; v < 8; ++v)
                Bf.u[v] = *(const unsigned int*)(wot + (size_t)(nt * 16 + l15) * HOUT + d0 +
                                                 2 * v + 16 * half);
            acc[nt] = bf16_wmma(A, Bf, acc[nt]);
        }
    }
#pragma unroll
    for (int nt = 0; nt < 4; ++nt)
#pragma unroll
        for (int r = 0; r < 8; ++r)
            out[(size_t)(row0 + r + 8 * half) * OUTD + nt * 16 + l15] = acc[nt][r];
}

// ---------------------------------------------------------------------------
// Host launcher
// ---------------------------------------------------------------------------
extern "C" void kernel_launch(void* const* d_in, const int* in_sizes, int n_in,
                              void* d_out, int out_size, void* d_ws, size_t ws_size,
                              hipStream_t stream) {
    const float* x  = (const float*)d_in[0];  // [B,S,D]
    const float* WQ = (const float*)d_in[1];  // [H,D,LAT]
    const float* WK = (const float*)d_in[2];  // [H,D,LAT]
    const float* WV = (const float*)d_in[3];  // [H,D,OUT]
    const float* WO = (const float*)d_in[4];  // [H*OUT,OUT]
    float* out = (float*)d_out;               // [B,S,OUT]

    // Workspace layout (bytes). Total ~81.6 MB.
    char* ws = (char*)d_ws;
    size_t off = 0;
    unsigned short* xb  = (unsigned short*)(ws + off); off += (size_t)BB * SS * DD * 2;     // 16 MB
    unsigned short* wqt = (unsigned short*)(ws + off); off += (size_t)HH * LATD * DD * 2;   // 512 KB
    unsigned short* wkt = (unsigned short*)(ws + off); off += (size_t)HH * LATD * DD * 2;
    unsigned short* wvt = (unsigned short*)(ws + off); off += (size_t)HH * OUTD * DD * 2;
    unsigned short* wot = (unsigned short*)(ws + off); off += (size_t)OUTD * HOUT * 2;      // 64 KB
    unsigned short* Qw  = (unsigned short*)(ws + off); off += (size_t)BB * HH * SS * LATD * 2;
    unsigned short* Kw  = (unsigned short*)(ws + off); off += (size_t)BB * HH * SS * LATD * 2;
    unsigned short* Vtw = (unsigned short*)(ws + off); off += (size_t)BB * HH * OUTD * SS * 2;
    unsigned short* Oc  = (unsigned short*)(ws + off); off += (size_t)BB * SS * HOUT * 2;
    (void)ws_size; (void)in_sizes; (void)n_in; (void)out_size;

    // 0) conversions / transposes
    {
        int n2 = BB * SS * DD / 2;  // 4,194,304 bf16 pairs
        cvt_x_kernel<<<(n2 + 255) / 256, 256, 0, stream>>>(x, (unsigned int*)xb, n2);
        int nw = HH * DD * LATD;    // 262,144
        transpose_bf16_kernel<<<(nw + 255) / 256, 256, 0, stream>>>(WQ, wqt, HH, DD, LATD);
        transpose_bf16_kernel<<<(nw + 255) / 256, 256, 0, stream>>>(WK, wkt, HH, DD, LATD);
        transpose_bf16_kernel<<<(nw + 255) / 256, 256, 0, stream>>>(WV, wvt, HH, DD, OUTD);
        int no = HOUT * OUTD;       // 32,768
        transpose_bf16_kernel<<<(no + 255) / 256, 256, 0, stream>>>(WO, wot, 1, HOUT, OUTD);
    }
    // 1) QKV projections: grid (B*H, S/128), 8 waves/block
    proj_kernel<<<dim3(BB * HH, SS / 128), 256, 0, stream>>>(xb, wqt, wkt, wvt, Qw, Kw, Vtw);
    // 2) flash attention: grid (B*H, S/128), K/V staged via async-to-LDS
    flash_kernel<<<dim3(BB * HH, SS / 128), 256, 0, stream>>>(Qw, Kw, Vtw, Oc);
    // 3) output projection: 1024 M-tiles, 8 per block
    outproj_kernel<<<(BB * SS / 16) / 8, 256, 0, stream>>>(Oc, wot, out);
}
